// RUNGLayer_combined_model_39505109188721
// MI455X (gfx1250) — matrix operand order repaired and checked by reading, not compiled, not run on hardware
//
#include <hip/hip_runtime.h>
#include <hip/hip_bf16.h>

#define N_NODES 4096
#define D_FEAT  256
#define EPS     1e-8f
#define SCAD_AA 3.7f
#define PCT_Q   0.75f
#define NBINS   2048
#define BIN_SCALE 1024.0f   // bins over [0,2], width = 1/1024

typedef __attribute__((ext_vector_type(16))) __bf16 v16bf;
typedef __attribute__((ext_vector_type(8)))  float  v8f;
typedef int v4i __attribute__((ext_vector_type(4)));
typedef __attribute__((address_space(1))) v4i* gp4;   // global v4i*
typedef __attribute__((address_space(3))) v4i* lp4;   // LDS v4i*

// ---- workspace layout (bytes) ----
#define WS_U_OFF    0
#define WS_HIST_OFF (N_NODES * D_FEAT * 2)
#define WS_LAM_OFF  (WS_HIST_OFF + NBINS * 4)

// ---------------------------------------------------------------------------
// CDNA5 async global->LDS copy (GLOBAL_LOAD_ASYNC_TO_LDS_B128, ASYNCcnt),
// guarded so the build still succeeds if the toolchain lacks the builtin.
// Signature (per hipcc diagnostic): (v4i global*, v4i lds*, imm off, imm cpol)
// ---------------------------------------------------------------------------
#if defined(__has_builtin)
#if __has_builtin(__builtin_amdgcn_global_load_async_to_lds_b128) && \
    __has_builtin(__builtin_amdgcn_s_wait_asynccnt)
#define HAVE_ASYNC_LDS 1
#endif
#endif

__device__ inline void cp16_async(void* lds_dst, const void* g_src) {
#ifdef HAVE_ASYNC_LDS
  __builtin_amdgcn_global_load_async_to_lds_b128((gp4)g_src, (lp4)lds_dst, 0, 0);
#else
  *(uint4*)lds_dst = *(const uint4*)g_src;
#endif
}
__device__ inline void cp_async_wait() {
#ifdef HAVE_ASYNC_LDS
  __builtin_amdgcn_s_wait_asynccnt(0);
#endif
}

static __device__ inline __bf16 f2bf(float x) {
  unsigned u = __float_as_uint(x);
  unsigned r = (u + 0x7FFFu + ((u >> 16) & 1u)) >> 16;   // round-to-nearest-even
  unsigned short s = (unsigned short)r;
  return __builtin_bit_cast(__bf16, s);
}

// ---------------------------------------------------------------------------
// Kernel 1: U[i,:] = (F[i,:]/Dsq[i]) / max(||.||, EPS), stored bf16
// ---------------------------------------------------------------------------
__global__ void k_normalize(const float* __restrict__ F, const float* __restrict__ Dsq,
                            __bf16* __restrict__ U) {
  __shared__ float red[D_FEAT];
  const int row = blockIdx.x;
  const int t = threadIdx.x;
  float v = F[(size_t)row * D_FEAT + t] / Dsq[row];
  red[t] = v * v;
  __syncthreads();
  for (int s = D_FEAT / 2; s > 0; s >>= 1) {
    if (t < s) red[t] += red[t + s];
    __syncthreads();
  }
  float nrm = fmaxf(sqrtf(red[0]), EPS);
  U[(size_t)row * D_FEAT + t] = f2bf(v / nrm);
}

// ---------------------------------------------------------------------------
// 16x16 cos-sim tile: 8x V_WMMA_F32_16X16X32_BF16 over K=256.
// A frag (16x32 bf16): lanes 0-15 row M=l hold K 0..7,16..23; lanes 16-31 +8.
// B frag: N=lane&15, K same as A frag. C/D: M = vgpr + 8*half, N = lane&15.
// ---------------------------------------------------------------------------
__device__ inline v8f wmma_cos_tile(const __bf16* __restrict__ sRow,
                                    const __bf16* __restrict__ sCol, int lane) {
  v8f c = {};
  const int half = lane >> 4;
  const int l = lane & 15;
  #pragma unroll
  for (int k0 = 0; k0 < D_FEAT; k0 += 32) {
    v16bf a, b;
    #pragma unroll
    for (int e = 0; e < 16; ++e) {
      const int p = e >> 1;
      const int ka = ((p < 4) ? (2 * p) : (8 + 2 * p)) + (e & 1) + 8 * half;
      a[e] = sRow[l * D_FEAT + k0 + ka];
      b[e] = sCol[l * D_FEAT + k0 + ((e < 8) ? e : (e + 8)) + 8 * half];
    }
    c = __builtin_amdgcn_wmma_f32_16x16x32_bf16(false, a, false, b,
                                                (short)0, c, false, false);
  }
  return c;
}

// ---------------------------------------------------------------------------
// Kernel 2: histogram of y over edges (A>0, i!=j). 8 waves = 8 col tiles.
// ---------------------------------------------------------------------------
__global__ void k_edge_hist(const __bf16* __restrict__ U, const float* __restrict__ A,
                            unsigned* __restrict__ hist) {
  __shared__ __align__(16) __bf16 sU[16 * D_FEAT];
  __shared__ __align__(16) __bf16 sV[128 * D_FEAT];
  __shared__ __align__(16) float  sA[16][128];
  const int I  = blockIdx.x >> 5;   // row tile 0..255
  const int Jg = blockIdx.x & 31;   // 128-col group 0..31
  const int t = threadIdx.x;

  const uint4* gU = (const uint4*)U;        // 32 uint4 per bf16 row
  const uint4* gA = (const uint4*)A;        // 1024 uint4 per f32 row
  uint4* lU = (uint4*)sU;
  uint4* lV = (uint4*)sV;
  uint4* lA = (uint4*)sA;
  for (int idx = t; idx < 16 * 32; idx += 256)
    cp16_async(lU + idx, gU + (size_t)I * 512 + idx);
  for (int idx = t; idx < 128 * 32; idx += 256)
    cp16_async(lV + idx, gU + (size_t)Jg * 4096 + idx);
  for (int idx = t; idx < 16 * 32; idx += 256)   // A tile rows I*16.., cols Jg*128..
    cp16_async(lA + idx,
               gA + (size_t)(I * 16 + (idx >> 5)) * 1024 + Jg * 32 + (idx & 31));
  cp_async_wait();
  __syncthreads();

  const int wave = t >> 5, lane = t & 31;
  const int half = lane >> 4, l = lane & 15;
  v8f c = wmma_cos_tile(sU, sV + wave * 16 * D_FEAT, lane);

  const int jl = wave * 16 + l;               // col within 128-group
  const int j = Jg * 128 + jl;
  #pragma unroll
  for (int r = 0; r < 8; ++r) {
    const int M = r + 8 * half;
    const int i = I * 16 + M;
    float y = fminf(fmaxf(1.0f - c[r], 0.0f), 2.0f);
    if (sA[M][jl] > 0.0f && i != j) {
      int bin = (int)(y * BIN_SCALE);
      bin = bin < 0 ? 0 : (bin > NBINS - 1 ? NBINS - 1 : bin);
      atomicAdd(&hist[bin], 1u);
    }
  }
}

// ---------------------------------------------------------------------------
// Kernel 3: approx 0.75-nanquantile via histogram -> lam_combined, gamma out
// ---------------------------------------------------------------------------
__global__ void k_quantile(const unsigned* __restrict__ hist,
                           const float* __restrict__ gamma_param_k,
                           const float* __restrict__ alpha_blend,
                           float* __restrict__ lam_out,
                           float* __restrict__ gamma_out) {
  unsigned total = 0;
  for (int b = 0; b < NBINS; ++b) total += hist[b];
  float gamma_data;
  if (total == 0) {
    gamma_data = EPS;
  } else {
    const float h = PCT_Q * (float)(total - 1);
    const unsigned k0 = (unsigned)h;
    const float frac = h - (float)k0;
    const unsigned k1 = (k0 + 1 < total) ? (k0 + 1) : k0;
    const float binw = 1.0f / BIN_SCALE;
    float v0 = 0.f, v1 = 0.f;
    bool got0 = false, got1 = false;
    unsigned cum = 0;
    for (int b = 0; b < NBINS && !(got0 && got1); ++b) {
      unsigned cnt = hist[b];
      if (cnt) {
        if (!got0 && k0 < cum + cnt) {
          v0 = ((float)b + ((float)(k0 - cum) + 0.5f) / (float)cnt) * binw; got0 = true;
        }
        if (!got1 && k1 < cum + cnt) {
          v1 = ((float)b + ((float)(k1 - cum) + 0.5f) / (float)cnt) * binw; got1 = true;
        }
        cum += cnt;
      }
    }
    gamma_data = fmaxf(v0 + frac * (v1 - v0), EPS);
  }
  const float lam_data  = gamma_data / SCAD_AA;
  const float lam_param = gamma_param_k[0] / SCAD_AA;
  const float alpha = alpha_blend[0];
  const float lam_c = alpha * lam_param + (1.0f - alpha) * lam_data;
  lam_out[0]   = lam_c;
  gamma_out[0] = lam_c * SCAD_AA;
}

// ---------------------------------------------------------------------------
// Kernel 4: per 16-row tile, software-pipelined over 32 j-groups:
//   [wmma y + SCAD -> wtile,qsum] | issue async stage(jg+1) | [sparse FMA] | wait
// ---------------------------------------------------------------------------
__global__ void k_fnew(const __bf16* __restrict__ U, const float* __restrict__ A,
                       const float* __restrict__ At, const float* __restrict__ Fc,
                       const float* __restrict__ F0, const float* __restrict__ Dv,
                       const float* __restrict__ lam_hat_p,
                       const float* __restrict__ lam_c_p,
                       float* __restrict__ Fnew) {
  __shared__ __align__(16) __bf16 sU[16 * D_FEAT];
  __shared__ __align__(16) __bf16 sV[128 * D_FEAT];
  __shared__ __align__(16) float  sA [16][128];
  __shared__ __align__(16) float  sAt[16][128];
  __shared__ float wtile[128][16];   // (W*A_tilde)[j_loc][i_loc]
  __shared__ float qsum[16];
  const int I = blockIdx.x;
  const int t = threadIdx.x;
  const int wave = t >> 5, lane = t & 31;
  const int half = lane >> 4, l = lane & 15;
  const float lam_c = lam_c_p[0];
  const float lam = 1.0f / lam_hat_p[0] - 1.0f;

  const uint4* gU  = (const uint4*)U;
  const uint4* gA  = (const uint4*)A;
  const uint4* gAt = (const uint4*)At;
  uint4* lU  = (uint4*)sU;
  uint4* lV  = (uint4*)sV;
  uint4* lA  = (uint4*)sA;
  uint4* lAt = (uint4*)sAt;

  // stage row tile + first j-group
  for (int idx = t; idx < 16 * 32; idx += 256)
    cp16_async(lU + idx, gU + (size_t)I * 512 + idx);
  {
    const int jg = 0;
    for (int idx = t; idx < 128 * 32; idx += 256)
      cp16_async(lV + idx, gU + (size_t)jg * 4096 + idx);
    for (int idx = t; idx < 16 * 32; idx += 256) {
      const size_t ro = (size_t)(I * 16 + (idx >> 5)) * 1024 + jg * 32 + (idx & 31);
      cp16_async(lA + idx, gA + ro);
      cp16_async(lAt + idx, gAt + ro);
    }
  }
  if (t < 16) qsum[t] = 0.0f;
  cp_async_wait();
  __syncthreads();

  const int i_loc = t >> 4;
  const int f0 = (t & 15) * 16;
  float acc[16];
  #pragma unroll
  for (int k = 0; k < 16; ++k) acc[k] = 0.0f;

  for (int jg = 0; jg < 32; ++jg) {
    // ---- phase 1: WMMA y tile + SCAD weights ----
    v8f c = wmma_cos_tile(sU, sV + wave * 16 * D_FEAT, lane);
    const int jl = wave * 16 + l;
    const int j = jg * 128 + jl;
    #pragma unroll
    for (int r = 0; r < 8; ++r) {
      const int M = r + 8 * half;
      const int i = I * 16 + M;
      float y = fminf(fmaxf(1.0f - c[r], 0.0f), 2.0f);
      float w;
      if (y <= lam_c)                 w = 1.0f;
      else if (y <= SCAD_AA * lam_c)  w = (SCAD_AA * lam_c - y) /
                                          ((SCAD_AA - 1.0f) * fmaxf(y, EPS));
      else                            w = 0.0f;
      if (i == j) w = 0.0f;
      wtile[jl][M] = w * sAt[M][jl];
      const float wa = w * sA[M][jl];
      if (wa != 0.0f) atomicAdd(&qsum[M], wa);
    }
    __syncthreads();   // wtile ready; sV/sA/sAt no longer needed

    // ---- issue async staging of next j-group (overlaps FMA phase) ----
    if (jg + 1 < 32) {
      const int ng = jg + 1;
      for (int idx = t; idx < 128 * 32; idx += 256)
        cp16_async(lV + idx, gU + (size_t)ng * 4096 + idx);
      for (int idx = t; idx < 16 * 32; idx += 256) {
        const size_t ro = (size_t)(I * 16 + (idx >> 5)) * 1024 + ng * 32 + (idx & 31);
        cp16_async(lA + idx, gA + ro);
        cp16_async(lAt + idx, gAt + ro);
      }
    }

    // ---- phase 2: sparse weighted feature accumulation ----
    const int jbase = jg * 128;
    for (int jj = 0; jj < 128; ++jj) {
      const float wa = wtile[jj][i_loc];
      if (wa != 0.0f) {              // ~32/4096 nonzero
        const float* fr = Fc + (size_t)(jbase + jj) * D_FEAT + f0;
        #pragma unroll
        for (int k = 0; k < 16; ++k) acc[k] += wa * fr[k];
      }
    }
    cp_async_wait();
    __syncthreads();
  }

  const int i = I * 16 + i_loc;
  const float Q = qsum[i_loc] / Dv[i] + lam;
  const float invQ = 1.0f / Q;
  #pragma unroll
  for (int k = 0; k < 16; ++k) {
    const float f0v = F0[(size_t)i * D_FEAT + f0 + k];
    Fnew[(size_t)i * D_FEAT + f0 + k] = acc[k] * invQ + lam * f0v * invQ;
  }
}

// ---------------------------------------------------------------------------
extern "C" void kernel_launch(void* const* d_in, const int* in_sizes, int n_in,
                              void* d_out, int out_size, void* d_ws, size_t ws_size,
                              hipStream_t stream) {
  const float* F   = (const float*)d_in[0];
  const float* F0  = (const float*)d_in[1];
  const float* A   = (const float*)d_in[2];
  const float* Dv  = (const float*)d_in[3];
  const float* Dsq = (const float*)d_in[4];
  const float* At  = (const float*)d_in[5];
  // d_in[6] = A_bool (unused: equivalent to A > 0)
  const float* lam_hat       = (const float*)d_in[7];
  const float* gamma_param_k = (const float*)d_in[8];
  const float* alpha_blend   = (const float*)d_in[9];

  __bf16*   U     = (__bf16*)  ((char*)d_ws + WS_U_OFF);
  unsigned* hist  = (unsigned*)((char*)d_ws + WS_HIST_OFF);
  float*    lam_c = (float*)   ((char*)d_ws + WS_LAM_OFF);

  float* Fnew      = (float*)d_out;
  float* gamma_out = Fnew + (size_t)N_NODES * D_FEAT;

  (void)hipMemsetAsync(hist, 0, NBINS * sizeof(unsigned), stream);
  k_normalize<<<N_NODES, D_FEAT, 0, stream>>>(F, Dsq, U);
  k_edge_hist<<<(N_NODES / 16) * (N_NODES / 128), 256, 0, stream>>>(U, A, hist);
  k_quantile<<<1, 1, 0, stream>>>(hist, gamma_param_k, alpha_blend, lam_c, gamma_out);
  k_fnew<<<N_NODES / 16, 256, 0, stream>>>(U, A, At, F, F0, Dv, lam_hat, lam_c, Fnew);
}
